// MultiheadAttention_35485019800104
// MI455X (gfx1250) — matrix operand order repaired and checked by reading, not compiled, not run on hardware
//
#include <hip/hip_runtime.h>

// ---------------------------------------------------------------------------
// MultiheadAttention forward for MI455X (gfx1250, wave32, WMMA bf16)
// BS=2, S=2048, D=1024, H=16, DH=64
// ---------------------------------------------------------------------------

typedef __attribute__((ext_vector_type(16))) __bf16 v16bf;
typedef __attribute__((ext_vector_type(8)))  __bf16 v8bf;
typedef __attribute__((ext_vector_type(8)))  float  v8f;

constexpr int BSZ = 2, S_ = 2048, D_ = 1024, H_ = 16, DH_ = 64;
constexpr int BM = 128, BN = 64, BK = 32;   // GEMM block tile

// f32 -> bf16: native cast lets the backend use the hardware convert path
// (RNE) instead of a manual bit-twiddle chain.
__device__ __forceinline__ __bf16 f2bf(float x) { return (__bf16)x; }

// --- gfx1250 async global->LDS copy (ASYNCcnt-tracked), 16 bytes ----------
// Builtin signature (from clang diagnostic): (global v4i*, LDS v4i*, Ii, Ii)
typedef int v4i __attribute__((vector_size(16)));
typedef __attribute__((address_space(1))) v4i gv4i;
typedef __attribute__((address_space(3))) v4i lv4i;

__device__ __forceinline__ void async_b128(const void* g, void* l) {
#if __has_builtin(__builtin_amdgcn_global_load_async_to_lds_b128)
  __builtin_amdgcn_global_load_async_to_lds_b128((gv4i*)g, (lv4i*)l, 0, 0);
#else
  asm volatile("global_load_async_to_lds_b128 %0, %1, off"
               :
               : "v"((unsigned)(unsigned long long)l), "v"(g)
               : "memory");
#endif
}

__device__ __forceinline__ void wait_async() {
#if __has_builtin(__builtin_amdgcn_s_wait_asynccnt)
  __builtin_amdgcn_s_wait_asynccnt(0);
#else
  asm volatile("s_wait_asynccnt 0" ::: "memory");
#endif
}

// A-matrix fragment (16xK=32 bf16), per ISA 7.12.2:
// lanes 0-15: row M=lane, K 0-7 (v0-3) and 16-23 (v4-7)
// lanes 16-31: row M=lane-16, K 8-15 and 24-31
__device__ __forceinline__ v16bf frag_a(const __bf16* base, int stride, int lane) {
  const int r = lane & 15, hi = (lane >> 4) & 1;
  const __bf16* p = base + r * stride;
  v8bf lo = *(const v8bf*)(p + hi * 8);
  v8bf hv = *(const v8bf*)(p + 16 + hi * 8);
  v16bf f;
#pragma unroll
  for (int i = 0; i < 8; ++i) { f[i] = lo[i]; f[i + 8] = hv[i]; }
  return f;
}

// B-matrix fragment (K=32 x 16 bf16): lane holds column N=lane%16,
// lanes 0-15 hold K 0-7 and 16-23, lanes 16-31 hold K 8-15 and 24-31.
__device__ __forceinline__ v16bf frag_b(const __bf16* base, int stride, int lane) {
  const int r = lane & 15, hi = (lane >> 4) & 1;
  const __bf16* p = base + r * stride + hi * 8;
  v8bf lo = *(const v8bf*)(p);
  v8bf hv = *(const v8bf*)(p + 16);
  v16bf f;
#pragma unroll
  for (int i = 0; i < 8; ++i) { f[i] = lo[i]; f[i + 8] = hv[i]; }
  return f;
}

__device__ __forceinline__ v8f wmma_bf16(v16bf a, v16bf b, v8f c) {
  return __builtin_amdgcn_wmma_f32_16x16x32_bf16(false, a, false, b, (short)0, c,
                                                 false, false);
}

// ---------------------------------------------------------------------------
// Kernel 1: fused QKV projection.  Y = X @ W^T + bias  (Linear semantics)
// X: (BS*S) x D f32 row-major, W: D x D f32 row-major (N x K)
// Output scattered bf16 into (b, h, s, dh); Q pre-scaled by 1/sqrt(DH).
// blockIdx.z selects Q/K/V.
// ---------------------------------------------------------------------------
__global__ __launch_bounds__(256) void qkv_proj_kernel(
    const float* __restrict__ Xq, const float* __restrict__ Xk,
    const float* __restrict__ Xv, const float* __restrict__ Wq,
    const float* __restrict__ Wk, const float* __restrict__ Wv,
    const float* __restrict__ bq, const float* __restrict__ bk,
    const float* __restrict__ bv, __bf16* __restrict__ Qo,
    __bf16* __restrict__ Ko, __bf16* __restrict__ Vo) {
  const int which = blockIdx.z;
  const float* X = which == 0 ? Xq : (which == 1 ? Xk : Xv);
  const float* W = which == 0 ? Wq : (which == 1 ? Wk : Wv);
  const float* bias = which == 0 ? bq : (which == 1 ? bk : bv);
  __bf16* Out = which == 0 ? Qo : (which == 1 ? Ko : Vo);
  const float scale = (which == 0) ? 0.125f : 1.0f;  // 1/sqrt(64) on Q

  __shared__ __bf16 sA[BM * BK];  // 8 KB
  __shared__ __bf16 sB[BN * BK];  // 4 KB

  const int tid = threadIdx.x;
  const int lane = tid & 31;
  const int wid = tid >> 5;  // 8 waves: 4 (M) x 2 (N)
  const int wm = wid >> 1, wn = wid & 1;
  const int mb = blockIdx.y * BM;
  const int nb = blockIdx.x * BN;

  const v8f z8 = {0.f, 0.f, 0.f, 0.f, 0.f, 0.f, 0.f, 0.f};
  v8f acc[2][2];
#pragma unroll
  for (int i = 0; i < 2; ++i)
#pragma unroll
    for (int j = 0; j < 2; ++j) acc[i][j] = z8;

  const int arow = tid >> 1, acol = (tid & 1) * 16;  // A: 128 rows, 16 f32 each
  const int brow = tid >> 2, bcol = (tid & 3) * 8;   // B: 64 rows, 8 f32 each

  for (int k0 = 0; k0 < D_; k0 += BK) {
    const float* ap = X + (size_t)(mb + arow) * D_ + k0 + acol;
#pragma unroll
    for (int i = 0; i < 16; i += 4) {
      float4 f = *(const float4*)(ap + i);
      __bf16* d = &sA[arow * BK + acol + i];
      d[0] = f2bf(f.x); d[1] = f2bf(f.y); d[2] = f2bf(f.z); d[3] = f2bf(f.w);
    }
    const float* bp = W + (size_t)(nb + brow) * D_ + k0 + bcol;
#pragma unroll
    for (int i = 0; i < 8; i += 4) {
      float4 f = *(const float4*)(bp + i);
      __bf16* d = &sB[brow * BK + bcol + i];
      d[0] = f2bf(f.x); d[1] = f2bf(f.y); d[2] = f2bf(f.z); d[3] = f2bf(f.w);
    }
    if (k0 + BK < D_) {  // hint next K tile into cache (global_prefetch_b8)
      __builtin_prefetch(ap + BK, 0, 0);
      __builtin_prefetch(bp + BK, 0, 0);
    }
    __syncthreads();

    v16bf a0 = frag_a(&sA[(wm * 32 + 0) * BK], BK, lane);
    v16bf a1 = frag_a(&sA[(wm * 32 + 16) * BK], BK, lane);
    v16bf b0 = frag_b(&sB[(wn * 32 + 0) * BK], BK, lane);
    v16bf b1 = frag_b(&sB[(wn * 32 + 16) * BK], BK, lane);
    acc[0][0] = wmma_bf16(a0, b0, acc[0][0]);
    acc[0][1] = wmma_bf16(a0, b1, acc[0][1]);
    acc[1][0] = wmma_bf16(a1, b0, acc[1][0]);
    acc[1][1] = wmma_bf16(a1, b1, acc[1][1]);
    __syncthreads();
  }

  // Epilogue: C layout -> lane r holds column, vgpr v holds row (lane half +8)
  const int r = lane & 15, hi = lane >> 4;
#pragma unroll
  for (int i = 0; i < 2; ++i)
#pragma unroll
    for (int j = 0; j < 2; ++j) {
      const int n = nb + wn * 32 + j * 16 + r;
      const float bv_ = bias[n];
      const int h = n >> 6, dh = n & 63;
#pragma unroll
      for (int v = 0; v < 8; ++v) {
        const int m = mb + wm * 32 + i * 16 + v + hi * 8;
        const int b = m >> 11, s = m & (S_ - 1);
        const float val = (acc[i][j][v] + bv_) * scale;
        Out[(((size_t)(b * H_ + h)) * S_ + s) * DH_ + dh] = f2bf(val);
      }
    }
}

// ---------------------------------------------------------------------------
// Kernel 2: flash-style attention with post-softmax multiplicative mask.
// Q/K/V bf16 in (b,h,s,dh).  4 waves/block, each wave = one 16-row q tile,
// block shares 32x64 K tile (async-copied) and transposed 64x32 V tile in LDS.
// context written bf16 as (b, s, h*DH+dh) for the output projection.
// ---------------------------------------------------------------------------
__global__ __launch_bounds__(128) void attn_kernel(
    const __bf16* __restrict__ Qb, const __bf16* __restrict__ Kb,
    const __bf16* __restrict__ Vb, const unsigned char* __restrict__ kpm,
    const float* __restrict__ amask, __bf16* __restrict__ Cb) {
  __shared__ __bf16 sK[32 * 64];      // K block: [kk][d]
  __shared__ __bf16 sV[64 * 32];      // V block transposed: [d][kk]
  __shared__ __bf16 sP[4][16 * 32];   // per-wave P relayout scratch

  const int tid = threadIdx.x;
  const int lane = tid & 31;
  const int w = tid >> 5;                 // wave id 0..3
  const int bh = blockIdx.y;              // b*H + h
  const int b = bh >> 4, h = bh & 15;
  const int qBase = blockIdx.x * 64 + w * 16;
  const int r = lane & 15, hi = lane >> 4;

  const __bf16* Qhead = Qb + (size_t)bh * S_ * DH_;
  const __bf16* Khead = Kb + (size_t)bh * S_ * DH_;
  const __bf16* Vhead = Vb + (size_t)bh * S_ * DH_;

  // Q fragments (A-matrix), two K=32 chunks over DH=64; loaded once.
  v16bf qf[2];
#pragma unroll
  for (int c = 0; c < 2; ++c)
    qf[c] = frag_a(Qhead + (size_t)qBase * DH_ + c * 32, DH_, lane);

  const v8f z8 = {0.f, 0.f, 0.f, 0.f, 0.f, 0.f, 0.f, 0.f};
  float mrun[8], lrun[8];
  v8f accd[4];
#pragma unroll
  for (int v = 0; v < 8; ++v) { mrun[v] = -1e30f; lrun[v] = 0.f; }
#pragma unroll
  for (int dt = 0; dt < 4; ++dt) accd[dt] = z8;

  const int krow = tid >> 2;          // 0..31
  const int kseg = (tid & 3) * 16;    // 0,16,32,48

  for (int kb = 0; kb < S_; kb += 32) {
    // --- stage K via async global->LDS DMA; V staged with transpose ---
    {
      const __bf16* srck = Khead + (size_t)(kb + krow) * DH_ + kseg;
      async_b128(srck, &sK[krow * 64 + kseg]);
      async_b128(srck + 8, &sK[krow * 64 + kseg + 8]);

      const __bf16* srcv = Vhead + (size_t)(kb + krow) * DH_ + kseg;
      v8bf y0 = *(const v8bf*)(srcv);
      v8bf y1 = *(const v8bf*)(srcv + 8);
#pragma unroll
      for (int e = 0; e < 8; ++e) {
        sV[(kseg + e) * 32 + krow] = y0[e];
        sV[(kseg + 8 + e) * 32 + krow] = y1[e];
      }
      if (kb + 32 < S_) __builtin_prefetch(srcv + 32 * DH_, 0, 0);
    }
    wait_async();     // barrier does not cover ASYNCcnt
    __syncthreads();

    // --- scores S = q @ k^T  (two 16-key tiles, K-dim = DH via 2 chunks) ---
    v8f sc[2] = {z8, z8};
#pragma unroll
    for (int t = 0; t < 2; ++t)
#pragma unroll
      for (int c = 0; c < 2; ++c) {
        v16bf bkf = frag_b(&sK[(t * 16) * 64 + c * 32], 64, lane);
        sc[t] = wmma_bf16(qf[c], bkf, sc[t]);
      }

    // key_padding_mask: invalid keys -> -inf (as large negative)
    const float km0 = kpm[(size_t)b * S_ + kb + r] ? 0.f : -1e30f;
    const float km1 = kpm[(size_t)b * S_ + kb + 16 + r] ? 0.f : -1e30f;

    // --- online softmax (row-wise over lanes of each 16-lane half) ---
#pragma unroll
    for (int v = 0; v < 8; ++v) {
      float s0 = sc[0][v] + km0;
      float s1 = sc[1][v] + km1;
      float tmax = fmaxf(s0, s1);
#pragma unroll
      for (int off = 1; off < 16; off <<= 1)
        tmax = fmaxf(tmax, __shfl_xor(tmax, off, 32));
      const float mnew = fmaxf(mrun[v], tmax);
      const float rescale = __expf(mrun[v] - mnew);
      mrun[v] = mnew;
      lrun[v] *= rescale;
#pragma unroll
      for (int dt = 0; dt < 4; ++dt) accd[dt][v] *= rescale;
      const float p0 = __expf(s0 - mnew);
      const float p1 = __expf(s1 - mnew);
      float psum = p0 + p1;
#pragma unroll
      for (int off = 1; off < 16; off <<= 1) psum += __shfl_xor(psum, off, 32);
      lrun[v] += psum;  // normalizer does NOT see attn_mask (post-softmax mask)

      // post-softmax multiplicative attn_mask, then re-layout P as A-matrix
      const int qrow = qBase + v + hi * 8;
      const float am0 = amask[((size_t)b * S_ + qrow) * S_ + kb + r];
      const float am1 = amask[((size_t)b * S_ + qrow) * S_ + kb + 16 + r];
      sP[w][(v + hi * 8) * 32 + r] = f2bf(p0 * am0);
      sP[w][(v + hi * 8) * 32 + 16 + r] = f2bf(p1 * am1);
    }

    // --- PV: acc += P(16x32) @ V(32x64) ---
    v16bf pf = frag_a(&sP[w][0], 32, lane);
#pragma unroll
    for (int dt = 0; dt < 4; ++dt) {
      v16bf bvf = frag_b(&sV[(dt * 16) * 32], 32, lane);
      accd[dt] = wmma_bf16(pf, bvf, accd[dt]);
    }
    __syncthreads();
  }

  // --- normalize and write context as (b, s, h*DH + dh) bf16 ---
#pragma unroll
  for (int dt = 0; dt < 4; ++dt)
#pragma unroll
    for (int v = 0; v < 8; ++v) {
      const int row = qBase + v + hi * 8;
      const int d = dt * 16 + r;
      const float val = accd[dt][v] / lrun[v];
      Cb[((size_t)(b * S_ + row)) * D_ + h * DH_ + d] = f2bf(val);
    }
}

// ---------------------------------------------------------------------------
// Kernel 3: output projection.  out = ctx @ Wo^T + bo  (f32 out)
// ctx bf16 (BS*S) x D row-major (async-copied to LDS), Wo f32 D x D (N x K).
// ---------------------------------------------------------------------------
__global__ __launch_bounds__(256) void out_proj_kernel(
    const __bf16* __restrict__ Cb, const float* __restrict__ Wo,
    const float* __restrict__ bo, float* __restrict__ Out) {
  __shared__ __bf16 sA[BM * BK];
  __shared__ __bf16 sB[BN * BK];

  const int tid = threadIdx.x;
  const int lane = tid & 31;
  const int wid = tid >> 5;
  const int wm = wid >> 1, wn = wid & 1;
  const int mb = blockIdx.y * BM;
  const int nb = blockIdx.x * BN;

  const v8f z8 = {0.f, 0.f, 0.f, 0.f, 0.f, 0.f, 0.f, 0.f};
  v8f acc[2][2];
#pragma unroll
  for (int i = 0; i < 2; ++i)
#pragma unroll
    for (int j = 0; j < 2; ++j) acc[i][j] = z8;

  const int arow = tid >> 1, acol = (tid & 1) * 16;
  const int brow = tid >> 2, bcol = (tid & 3) * 8;

  for (int k0 = 0; k0 < D_; k0 += BK) {
    const __bf16* ap = Cb + (size_t)(mb + arow) * D_ + k0 + acol;
    async_b128(ap, &sA[arow * BK + acol]);        // bf16 A tile: pure DMA
    async_b128(ap + 8, &sA[arow * BK + acol + 8]);

    const float* bp = Wo + (size_t)(nb + brow) * D_ + k0 + bcol;
#pragma unroll
    for (int i = 0; i < 8; i += 4) {
      float4 f = *(const float4*)(bp + i);
      __bf16* d = &sB[brow * BK + bcol + i];
      d[0] = f2bf(f.x); d[1] = f2bf(f.y); d[2] = f2bf(f.z); d[3] = f2bf(f.w);
    }
    if (k0 + BK < D_) {
      __builtin_prefetch(ap + BK, 0, 0);
      __builtin_prefetch(bp + BK, 0, 0);
    }
    wait_async();
    __syncthreads();

    v16bf a0 = frag_a(&sA[(wm * 32 + 0) * BK], BK, lane);
    v16bf a1 = frag_a(&sA[(wm * 32 + 16) * BK], BK, lane);
    v16bf b0 = frag_b(&sB[(wn * 32 + 0) * BK], BK, lane);
    v16bf b1 = frag_b(&sB[(wn * 32 + 16) * BK], BK, lane);
    acc[0][0] = wmma_bf16(a0, b0, acc[0][0]);
    acc[0][1] = wmma_bf16(a0, b1, acc[0][1]);
    acc[1][0] = wmma_bf16(a1, b0, acc[1][0]);
    acc[1][1] = wmma_bf16(a1, b1, acc[1][1]);
    __syncthreads();
  }

  const int r = lane & 15, hi = lane >> 4;
#pragma unroll
  for (int i = 0; i < 2; ++i)
#pragma unroll
    for (int j = 0; j < 2; ++j) {
      const int n = nb + wn * 32 + j * 16 + r;
      const float bv_ = bo[n];
#pragma unroll
      for (int v = 0; v < 8; ++v) {
        const int m = mb + wm * 32 + i * 16 + v + hi * 8;
        Out[(size_t)m * D_ + n] = acc[i][j][v] + bv_;
      }
    }
}

// ---------------------------------------------------------------------------
// launch
// ---------------------------------------------------------------------------
extern "C" void kernel_launch(void* const* d_in, const int* in_sizes, int n_in,
                              void* d_out, int out_size, void* d_ws,
                              size_t ws_size, hipStream_t stream) {
  const float* query = (const float*)d_in[0];
  const float* key = (const float*)d_in[1];
  const float* value = (const float*)d_in[2];
  const unsigned char* kpm = (const unsigned char*)d_in[3];
  const float* amask = (const float*)d_in[4];
  const float* Wq = (const float*)d_in[5];
  const float* bq = (const float*)d_in[6];
  const float* Wk = (const float*)d_in[7];
  const float* bk = (const float*)d_in[8];
  const float* Wv = (const float*)d_in[9];
  const float* bv = (const float*)d_in[10];
  const float* Wo = (const float*)d_in[11];
  const float* bo = (const float*)d_in[12];

  const size_t per = (size_t)BSZ * S_ * D_;  // elements per bf16 buffer
  __bf16* Qb = (__bf16*)d_ws;
  __bf16* Kb = Qb + per;
  __bf16* Vb = Kb + per;
  __bf16* Cb = Vb + per;

  dim3 gProj(D_ / BN, (BSZ * S_) / BM, 3);  // (16, 32, 3)
  qkv_proj_kernel<<<gProj, 256, 0, stream>>>(query, key, value, Wq, Wk, Wv, bq,
                                             bk, bv, Qb, Kb, Vb);

  dim3 gAttn(S_ / 64, BSZ * H_);  // (32, 32)
  attn_kernel<<<gAttn, 128, 0, stream>>>(Qb, Kb, Vb, kpm, amask, Cb);

  dim3 gOut(D_ / BN, (BSZ * S_) / BM);  // (16, 32)
  out_proj_kernel<<<gOut, 256, 0, stream>>>(Cb, Wo, bo, (float*)d_out);
}
